// DualGATEncoder_2731599200598
// MI455X (gfx1250) — matrix operand order repaired and checked by reading, not compiled, not run on hardware
//
#include <hip/hip_runtime.h>
#include <hip/hip_bf16.h>
#include <math.h>

// ---------------------------------------------------------------------------
// Types for CDNA5 WMMA (wave32): v_wmma_f32_16x16x32_bf16
// ---------------------------------------------------------------------------
typedef __bf16 bf16_t;
typedef __attribute__((ext_vector_type(16))) __bf16 v16bf;
typedef __attribute__((ext_vector_type(8)))  float  v8f;

#define BM 128
#define BN 128
#define BK 32
#define LDS_PAD 8
#define LDS_LD (BK + LDS_PAD)   // 40 bf16 per row -> 80B row stride (16B aligned)

union FragU { v16bf v; uint4 q[2]; };

__device__ __forceinline__ unsigned pkbf16(float a, float b) {
    union { bf16_t h[2]; unsigned u; } p;
    p.h[0] = (bf16_t)a; p.h[1] = (bf16_t)b;
    return p.u;
}

// ---------------------------------------------------------------------------
// Generic GEMM:  C[m, coff+n] = act(A[m,:K] @ W[:K, nBase..] + bias[n])
// A: f32 row-major (lda), W: f32 row-major (ldw = total N).  Tiles converted
// to bf16 in LDS; double-buffered with register staging so next-tile global
// loads are issued before the WMMA burst and stored after it.
// Workgroup = 128x128 tile, 8 waves in a 4x2 grid: each wave owns 32 rows x
// 64 cols = 2 A-frags x 4 B-frags -> 8 WMMAs per K-step from 12 DS reads.
// act: 0 = identity, 1 = exact GELU.
// ---------------------------------------------------------------------------
__global__ __launch_bounds__(256)
void gemm_bias_act(const float* __restrict__ A, int lda,
                   const float* __restrict__ W, int ldw,
                   const float* __restrict__ bias,
                   float* __restrict__ C, int ldc, int coff,
                   int M, int K, int act)
{
    __shared__ bf16_t sA[2][BM][LDS_LD];          // [m][k]
    __shared__ bf16_t sB[2][BN][LDS_LD];          // transposed: [n][k]

    const int tid  = threadIdx.x;
    const int lane = tid & 31;
    const int wave = tid >> 5;
    const int wr   = wave >> 1;                   // row group: 32 rows each
    const int wc   = wave & 1;                    // col group: 64 cols each
    const int mBase = blockIdx.x * BM;
    const int nBase = blockIdx.y * BN;

    v8f acc[2][4];
    #pragma unroll
    for (int i = 0; i < 2; ++i)
        #pragma unroll
        for (int j = 0; j < 4; ++j)
            #pragma unroll
            for (int k = 0; k < 8; ++k) acc[i][j][k] = 0.f;

    const int nk = K / BK;

    // ---- prologue: stage tile 0 straight into LDS buffer 0 ----
    {
        #pragma unroll
        for (int it = 0; it < 4; ++it) {
            int f = tid + it * 256;             // float4 id 0..1023
            int r = f >> 3;                     // row 0..127
            int c = (f & 7) << 2;               // col 0,4,..,28
            float4 v = make_float4(0.f, 0.f, 0.f, 0.f);
            int gm = mBase + r;
            if (gm < M) v = *(const float4*)(A + (size_t)gm * lda + c);
            uint2 p; p.x = pkbf16(v.x, v.y); p.y = pkbf16(v.z, v.w);
            *(uint2*)&sA[0][r][c] = p;
        }
        #pragma unroll
        for (int it = 0; it < 4; ++it) {
            int f  = tid + it * 256;            // (n, k4) id 0..1023
            int n  = f & 127;
            int k4 = (f >> 7) << 2;             // 0,4,..,28
            const float* wp = W + (size_t)k4 * ldw + nBase + n;
            float4 v;
            v.x = wp[0]; v.y = wp[ldw]; v.z = wp[2 * (size_t)ldw]; v.w = wp[3 * (size_t)ldw];
            uint2 p; p.x = pkbf16(v.x, v.y); p.y = pkbf16(v.z, v.w);
            *(uint2*)&sB[0][n][k4] = p;
        }
    }
    __syncthreads();

    for (int ki = 0; ki < nk; ++ki) {
        const int buf = ki & 1;
        const bool pre = (ki + 1 < nk);

        // ---- 1. issue next tile's global loads into registers (no wait) ----
        float4 ra[4], rb[4];
        if (pre) {
            int kn = (ki + 1) * BK;
            #pragma unroll
            for (int it = 0; it < 4; ++it) {
                int f = tid + it * 256;
                int r = f >> 3;
                int c = (f & 7) << 2;
                ra[it] = make_float4(0.f, 0.f, 0.f, 0.f);
                int gm = mBase + r;
                if (gm < M) ra[it] = *(const float4*)(A + (size_t)gm * lda + kn + c);
            }
            #pragma unroll
            for (int it = 0; it < 4; ++it) {
                int f  = tid + it * 256;
                int n  = f & 127;
                int k4 = (f >> 7) << 2;
                const float* wp = W + (size_t)(kn + k4) * ldw + nBase + n;
                rb[it].x = wp[0];
                rb[it].y = wp[ldw];
                rb[it].z = wp[2 * (size_t)ldw];
                rb[it].w = wp[3 * (size_t)ldw];
            }
            // L2 prefetch two tiles ahead (global_prefetch_b8)
            if (ki + 2 < nk) {
                int gm = mBase + (tid >> 1);
                if (gm < M)
                    __builtin_prefetch(A + (size_t)gm * lda + (ki + 2) * BK, 0, 1);
            }
        }

        // ---- 2. fragment loads (batched), then WMMA burst ----
        // A fragment (16x32 bf16, ISA 7.12.2): lanes 0-15 M=lane K{0..7,16..23},
        // lanes 16-31 M=lane-16 K{8..15,24..31}  -> khalf = (lane&16)>>1
        FragU ua[2];
        #pragma unroll
        for (int ms = 0; ms < 2; ++ms) {
            const bf16_t* arow =
                &sA[buf][wr * 32 + ms * 16 + (lane & 15)][(lane & 16) >> 1];
            ua[ms].q[0] = *(const uint4*)(arow);
            ua[ms].q[1] = *(const uint4*)(arow + 16);
        }
        FragU ub[4];
        #pragma unroll
        for (int nf = 0; nf < 4; ++nf) {
            const bf16_t* bcol =
                &sB[buf][wc * 64 + nf * 16 + (lane & 15)][(lane & 16) >> 1];
            ub[nf].q[0] = *(const uint4*)(bcol);
            ub[nf].q[1] = *(const uint4*)(bcol + 16);
        }
        #pragma unroll
        for (int nf = 0; nf < 4; ++nf)
            #pragma unroll
            for (int ms = 0; ms < 2; ++ms)
                acc[ms][nf] = __builtin_amdgcn_wmma_f32_16x16x32_bf16(
                    false, ua[ms].v, false, ub[nf].v, (short)0, acc[ms][nf],
                    false, false);
#if __has_builtin(__builtin_amdgcn_sched_group_barrier)
        // Pin the shape: 12 DS reads first, then 8 WMMAs (DScnt retires in
        // order, so partial waits overlap remaining loads with matrix issue).
        __builtin_amdgcn_sched_group_barrier(0x100, 12, 0);  // DS read x12
        __builtin_amdgcn_sched_group_barrier(0x008, 8, 0);   // MFMA/WMMA x8
#endif

        // ---- 3. convert + store the staged tile (loadcnt wait lands here) ----
        if (pre) {
            const int nb = buf ^ 1;
            #pragma unroll
            for (int it = 0; it < 4; ++it) {
                int f = tid + it * 256;
                int r = f >> 3;
                int c = (f & 7) << 2;
                uint2 p; p.x = pkbf16(ra[it].x, ra[it].y); p.y = pkbf16(ra[it].z, ra[it].w);
                *(uint2*)&sA[nb][r][c] = p;
            }
            #pragma unroll
            for (int it = 0; it < 4; ++it) {
                int f  = tid + it * 256;
                int n  = f & 127;
                int k4 = (f >> 7) << 2;
                uint2 p; p.x = pkbf16(rb[it].x, rb[it].y); p.y = pkbf16(rb[it].z, rb[it].w);
                *(uint2*)&sB[nb][n][k4] = p;
            }
        }
        __syncthreads();
    }

    // ---- epilogue: C/D layout — n = lane&15, m = v + 8*(lane>=16) ----
    #pragma unroll
    for (int nf = 0; nf < 4; ++nf) {
        int gn = nBase + wc * 64 + nf * 16 + (lane & 15);
        float b = bias[gn];
        #pragma unroll
        for (int ms = 0; ms < 2; ++ms) {
            #pragma unroll
            for (int v = 0; v < 8; ++v) {
                int gm = mBase + wr * 32 + ms * 16 + ((lane & 16) >> 1) + v;
                if (gm < M) {
                    float x = acc[ms][nf][v] + b;
                    if (act == 1)
                        x = 0.5f * x * (1.f + erff(x * 0.70710678118654752f));
                    C[(size_t)gm * ldc + coff + gn] = x;
                }
            }
        }
    }
}

// ---------------------------------------------------------------------------
// Monotone float<->uint key for atomic segment-max
// ---------------------------------------------------------------------------
__device__ __forceinline__ unsigned fkey(float x) {
    unsigned b = __float_as_uint(x);
    return (b & 0x80000000u) ? ~b : (b | 0x80000000u);
}
__device__ __forceinline__ float funkey(unsigned k) {
    unsigned b = (k & 0x80000000u) ? (k ^ 0x80000000u) : ~k;
    return __uint_as_float(b);
}
__device__ __forceinline__ float lrelu(float x) { return x > 0.f ? x : 0.2f * x; }

// ---------------------------------------------------------------------------
// Pass 1: wave-per-edge logits + atomic segment max.
// logits[e,h] = sum_c att[h,c] * leaky_relu(xl[s,h,c] + xr[d,h,c], 0.2)
// ---------------------------------------------------------------------------
__global__ __launch_bounds__(256)
void edge_logits(const float* __restrict__ xl, const float* __restrict__ xr,
                 const int* __restrict__ src, const int* __restrict__ dst,
                 int E, int Etot, const float* __restrict__ att,
                 int H, int C, float* __restrict__ logits,
                 unsigned* __restrict__ mkey)
{
    int e    = (blockIdx.x * blockDim.x + threadIdx.x) >> 5;
    int lane = threadIdx.x & 31;
    if (e >= Etot) return;
    int s = (e < E) ? src[e] : (e - E);
    int d = (e < E) ? dst[e] : (e - E);

    int HC    = H * C;
    int chunk = HC >> 5;                 // 16 (H=4) or 4 (H=1); within one head
    int i0    = lane * chunk;
    int h     = i0 / C;

    const float* pl = xl  + (size_t)s * HC + i0;
    const float* pr = xr  + (size_t)d * HC + i0;
    const float* pa = att + i0;          // att is [H][C] flat -> index == i0

    float sum = 0.f;
    for (int j = 0; j < chunk; j += 4) {
        float4 a = *(const float4*)(pl + j);
        float4 b = *(const float4*)(pr + j);
        float4 w = *(const float4*)(pa + j);
        sum += w.x * lrelu(a.x + b.x);
        sum += w.y * lrelu(a.y + b.y);
        sum += w.z * lrelu(a.z + b.z);
        sum += w.w * lrelu(a.w + b.w);
    }
    int lanesPerHead = C / chunk;        // 8 or 32
    for (int off = lanesPerHead >> 1; off > 0; off >>= 1)
        sum += __shfl_xor(sum, off, 32);
    if ((lane & (lanesPerHead - 1)) == 0) {
        logits[(size_t)e * H + h] = sum;
        atomicMax(&mkey[d * H + h], fkey(sum));
    }
}

// ---------------------------------------------------------------------------
// Pass 2: ex = exp(logit - max[dst]); den[dst] += ex  (overwrite logits w/ ex)
// ---------------------------------------------------------------------------
__global__ __launch_bounds__(256)
void edge_expsum(float* __restrict__ logits, const unsigned* __restrict__ mkey,
                 float* __restrict__ den, const int* __restrict__ dst,
                 int E, int Etot, int H)
{
    long t = (long)blockIdx.x * blockDim.x + threadIdx.x;
    if (t >= (long)Etot * H) return;
    int e = (int)(t / H);
    int h = (int)(t - (long)e * H);
    int d = (e < E) ? dst[e] : (e - E);
    float m  = funkey(mkey[d * H + h]);
    float ex = expf(logits[t] - m);
    logits[t] = ex;
    atomicAdd(&den[d * H + h], ex);
}

// ---------------------------------------------------------------------------
// Pass 3: wave-per-edge weighted scatter:  agg[d,:] += alpha_h * xl[s,:]
// ---------------------------------------------------------------------------
__global__ __launch_bounds__(256)
void edge_aggregate(const float* __restrict__ xl, const float* __restrict__ ex,
                    const float* __restrict__ den,
                    const int* __restrict__ src, const int* __restrict__ dst,
                    int E, int Etot, int H, int C, float* __restrict__ agg)
{
    int e    = (blockIdx.x * blockDim.x + threadIdx.x) >> 5;
    int lane = threadIdx.x & 31;
    if (e >= Etot) return;
    int s = (e < E) ? src[e] : (e - E);
    int d = (e < E) ? dst[e] : (e - E);

    int HC    = H * C;
    int chunk = HC >> 5;
    int i0    = lane * chunk;
    int h     = i0 / C;

    float alpha = ex[(size_t)e * H + h] / den[d * H + h];
    const float* pl = xl  + (size_t)s * HC + i0;
    float*       po = agg + (size_t)d * HC + i0;
    for (int j = 0; j < chunk; ++j)
        atomicAdd(&po[j], alpha * pl[j]);
}

// ---------------------------------------------------------------------------
// Pass 4: out[n, coff+i] = elu(agg[n,i] + bias[i])
// ---------------------------------------------------------------------------
__global__ __launch_bounds__(256)
void gat_finalize(const float* __restrict__ agg, const float* __restrict__ bias,
                  int N, int HC, float* __restrict__ out, int ldo, int coff)
{
    long t = (long)blockIdx.x * blockDim.x + threadIdx.x;
    if (t >= (long)N * HC) return;
    int n = (int)(t / HC);
    int i = (int)(t - (long)n * HC);
    float x = agg[t] + bias[i];
    out[(size_t)n * ldo + coff + i] = (x > 0.f) ? x : expm1f(x);
}

// ---------------------------------------------------------------------------
// Host orchestration
// ---------------------------------------------------------------------------
extern "C" void kernel_launch(void* const* d_in, const int* in_sizes, int n_in,
                              void* d_out, int out_size, void* d_ws, size_t ws_size,
                              hipStream_t stream)
{
    (void)n_in; (void)out_size; (void)ws_size;
    const float* packet = (const float*)d_in[0];
    const float* timef  = (const float*)d_in[1];
    const int*   eidx   = (const int*)d_in[2];       // int32 [2][E]

    struct Mlp { const float *W1, *b1, *W2, *b2; };
    struct Gat { const float *Wl, *bl, *Wr, *br, *att, *bias; };
    int idx = 3;
    auto nf = [&]() { return (const float*)d_in[idx++]; };
    Mlp pe  { nf(), nf(), nf(), nf() };
    Mlp te  { nf(), nf(), nf(), nf() };
    Gat pg1 { nf(), nf(), nf(), nf(), nf(), nf() };
    Gat pg2 { nf(), nf(), nf(), nf(), nf(), nf() };
    Gat tg1 { nf(), nf(), nf(), nf(), nf(), nf() };
    Gat tg2 { nf(), nf(), nf(), nf(), nf(), nf() };
    Mlp enh { nf(), nf(), nf(), nf() };

    const int N    = in_sizes[0] / 4096;
    const int E    = in_sizes[2] / 2;
    const int Etot = E + N;
    const int* src = eidx;
    const int* dstp = eidx + E;

    // ---- workspace carve (bump allocator, 256B aligned) ----
    char* w = (char*)d_ws;
    auto carve = [&](size_t bytes) {
        char* p = w; w += (bytes + 255) & ~(size_t)255; return p;
    };
    float*    xl    = (float*)carve((size_t)N * 512 * 4);
    float*    xr    = (float*)carve((size_t)N * 512 * 4);   // later reused as x1
    float*    agg   = (float*)carve((size_t)N * 512 * 4);
    float*    x0    = (float*)carve((size_t)N * 256 * 4);   // later xl2|xr2
    float*    hbuf  = (float*)carve((size_t)N * 512 * 4);   // h1 / henh
    float*    xcat  = (float*)carve((size_t)N * 256 * 4);
    float*    logit = (float*)carve((size_t)Etot * 4 * 4);
    unsigned* mkey  = (unsigned*)carve((size_t)N * 4 * 4);
    float*    den   = (float*)carve((size_t)N * 4 * 4);

    const int mt = (N + BM - 1) / BM;
    auto gemm = [&](const float* A, int lda, const float* W, int Nc,
                    const float* bias, float* C, int ldc, int coff,
                    int K, int act) {
        dim3 grid(mt, Nc / 128);
        gemm_bias_act<<<grid, 256, 0, stream>>>(A, lda, W, Nc, bias,
                                                C, ldc, coff, N, K, act);
    };
    const int ew = (Etot * 32 + 255) / 256;   // wave-per-edge grid

    for (int br = 0; br < 2; ++br) {
        const float* feat = br ? timef : packet;
        const Mlp& ext = br ? te  : pe;
        const Gat& g1  = br ? tg1 : pg1;
        const Gat& g2  = br ? tg2 : pg2;

        // feature extractor: 4096 -> 128 (GELU) -> 256
        gemm(feat, 4096, ext.W1, 128, ext.b1, hbuf, 128, 0, 4096, 1);
        gemm(hbuf,  128, ext.W2, 256, ext.b2, x0,   256, 0,  128, 0);

        // GATv2 layer 1 (H=4, C=128)
        gemm(x0, 256, g1.Wl, 512, g1.bl, xl, 512, 0, 256, 0);
        gemm(x0, 256, g1.Wr, 512, g1.br, xr, 512, 0, 256, 0);
        hipMemsetAsync(mkey, 0, (size_t)N * 4 * sizeof(unsigned), stream);
        hipMemsetAsync(den,  0, (size_t)N * 4 * sizeof(float), stream);
        hipMemsetAsync(agg,  0, (size_t)N * 512 * sizeof(float), stream);
        edge_logits<<<ew, 256, 0, stream>>>(xl, xr, src, dstp, E, Etot,
                                            g1.att, 4, 128, logit, mkey);
        edge_expsum<<<((size_t)Etot * 4 + 255) / 256, 256, 0, stream>>>(
            logit, mkey, den, dstp, E, Etot, 4);
        edge_aggregate<<<ew, 256, 0, stream>>>(xl, logit, den, src, dstp,
                                               E, Etot, 4, 128, agg);
        // x1 := elu(agg + bias), reuse xr buffer
        gat_finalize<<<((size_t)N * 512 + 255) / 256, 256, 0, stream>>>(
            agg, g1.bias, N, 512, xr, 512, 0);

        // GATv2 layer 2 (H=1, C=128); xl2/xr2 live in x0 region
        float* xl2 = x0;
        float* xr2 = x0 + (size_t)N * 128;
        gemm(xr, 512, g2.Wl, 128, g2.bl, xl2, 128, 0, 512, 0);
        gemm(xr, 512, g2.Wr, 128, g2.br, xr2, 128, 0, 512, 0);
        hipMemsetAsync(mkey, 0, (size_t)N * sizeof(unsigned), stream);
        hipMemsetAsync(den,  0, (size_t)N * sizeof(float), stream);
        hipMemsetAsync(agg,  0, (size_t)N * 128 * sizeof(float), stream);
        edge_logits<<<ew, 256, 0, stream>>>(xl2, xr2, src, dstp, E, Etot,
                                            g2.att, 1, 128, logit, mkey);
        edge_expsum<<<((size_t)Etot + 255) / 256, 256, 0, stream>>>(
            logit, mkey, den, dstp, E, Etot, 1);
        edge_aggregate<<<ew, 256, 0, stream>>>(xl2, logit, den, src, dstp,
                                               E, Etot, 1, 128, agg);
        // write branch result into concat buffer (col offset 0 / 128)
        gat_finalize<<<((size_t)N * 128 + 255) / 256, 256, 0, stream>>>(
            agg, g2.bias, N, 128, xcat, 256, br ? 128 : 0);
    }

    // enhancement MLP: 256 -> 512 (GELU) -> 256
    gemm(xcat, 256, enh.W1, 512, enh.b1, hbuf, 512, 0, 256, 1);
    gemm(hbuf, 512, enh.W2, 256, enh.b2, (float*)d_out, 256, 0, 512, 0);
}
